// MultiHeadDilatedLocalAttention_27685359190792
// MI455X (gfx1250) — compile-verified, matrix operand order checked by reading, not run on hardware
//
#include <hip/hip_runtime.h>

// ---------------------------------------------------------------------------
// MultiHeadDilatedLocalAttention on MI455X (gfx1250)
//   B=2, S=2048, D=1024, H=16, DH=64, WINDOW=257 (half=128), DILATION=2,
//   N_GLOBAL=32. Compute-bound (everything fits in 192MB L2) -> bf16 WMMA
//   (v_wmma_f32_16x16x32_bf16) with fp32 accumulation for all GEMMs and the
//   banded flash-attention. GEMM tile staging uses the gfx1250 async
//   global->LDS DMA path (global_load_async_to_lds_b128 + s_wait_asynccnt)
//   with statically double-buffered LDS (2x-unrolled K-loop, peeled tail).
// ---------------------------------------------------------------------------

#define S_LEN   2048
#define BATCH   2
#define DMODEL  1024
#define HEADS   16
#define DHEAD   64
#define HALF_WIN 128
#define NWORDS  (S_LEN / 32)

typedef __bf16 v16bf  __attribute__((ext_vector_type(16)));
typedef __bf16 bf16x8 __attribute__((ext_vector_type(8)));
typedef __bf16 bf16x4 __attribute__((ext_vector_type(4)));
typedef float  v8f    __attribute__((ext_vector_type(8)));

union FragAB { v16bf v; bf16x8 h[2]; };

__device__ inline v8f wmma_bf16(v16bf a, v16bf b, v8f c) {
  return __builtin_amdgcn_wmma_f32_16x16x32_bf16(false, a, false, b,
                                                 (short)0, c, false, false);
}

// Byte offset of a generic pointer within the wave's LDS aperture.
__device__ inline unsigned lds_byte_off(void* p) {
  return (unsigned)(uintptr_t)(void __attribute__((address_space(3)))*)p;
}

// gfx1250 async DMA: global memory -> LDS, 16 bytes per lane, ASYNCcnt.
__device__ inline void async_ld_b128(unsigned ldsoff, const void* g) {
  asm volatile("global_load_async_to_lds_b128 %0, %1, off"
               :: "v"(ldsoff), "v"((unsigned long long)(uintptr_t)g)
               : "memory");
}

// ---------------------------------------------------------------------------
// fp32 -> bf16 convert (n multiple of 4)
// ---------------------------------------------------------------------------
__global__ void cvt_f32_bf16(const float* __restrict__ in,
                             __bf16* __restrict__ out, int n4) {
  int i = blockIdx.x * blockDim.x + threadIdx.x;
  int stride = gridDim.x * blockDim.x;
  for (; i < n4; i += stride) {
    float4 f = ((const float4*)in)[i];
    bf16x4 o;
    o[0] = (__bf16)f.x; o[1] = (__bf16)f.y;
    o[2] = (__bf16)f.z; o[3] = (__bf16)f.w;
    ((bf16x4*)out)[i] = o;
  }
}

// ---------------------------------------------------------------------------
// Build per-position global-token bitmask from global_idx (int64)
// ---------------------------------------------------------------------------
__global__ void build_gmask(const long long* __restrict__ gidx,
                            unsigned* __restrict__ words, int n_glob) {
  int t = threadIdx.x;
  if (t < NWORDS) words[t] = 0u;
  __syncthreads();
  if (t < n_glob) {
    long long g = gidx[t];
    atomicOr(&words[(int)(g >> 5)], 1u << ((int)g & 31));
  }
}

// ---------------------------------------------------------------------------
// GEMM: C[M,N] = A[M,K] (bf16) @ W[N,K]^T (bf16) + bias (fp32)
//   block = 128x128, K-step 32, 8 waves (wave computes 32x64 = 2x4 C tiles)
//   Statically double-buffered LDS filled by async global->LDS DMA.
// ---------------------------------------------------------------------------
#define BM 128
#define BN 128
#define BK 32
#define LSTR 40              // padded LDS row stride in halves (80B, 16B-mult)
#define LBUF (BM * LSTR)     // halves per operand per buffer (10240 B)

__global__ __launch_bounds__(256) void gemm_bf16(
    const __bf16* __restrict__ A, const __bf16* __restrict__ W,
    const float* __restrict__ bias, __bf16* __restrict__ outb,
    float* __restrict__ outf, int M, int N, int K) {
  __shared__ __attribute__((aligned(16))) __bf16 lA[2 * LBUF];
  __shared__ __attribute__((aligned(16))) __bf16 lB[2 * LBUF];

  const int t    = threadIdx.x;
  const int lane = t & 31;
  const int wvid = t >> 5;
  const int col  = lane & 15;
  const int hl   = (lane >> 4) & 1;
  const int waveM = wvid >> 1;   // 0..3 -> 32-row slice
  const int waveN = wvid & 1;    // 0..1 -> 64-col slice
  const int m0 = blockIdx.y * BM;
  const int n0 = blockIdx.x * BN;

  v8f acc[2][4] = {};

  const int ldr = t >> 2;         // 0..63
  const int ldc = (t & 3) * 8;    // halves within the 32-wide k slab

  // per-thread LDS destination byte addresses, both buffers, both tile rows
  unsigned dstA[2][2], dstB[2][2];
  {
    const unsigned aBase = lds_byte_off(lA);
    const unsigned bBase = lds_byte_off(lB);
    #pragma unroll
    for (int bsel = 0; bsel < 2; ++bsel) {
      #pragma unroll
      for (int rr = 0; rr < 2; ++rr) {
        const unsigned off =
            (unsigned)((ldr + rr * 64) * LSTR + ldc) * 2 +
            (unsigned)bsel * (LBUF * 2);
        dstA[bsel][rr] = aBase + off;
        dstB[bsel][rr] = bBase + off;
      }
    }
  }
  const __bf16* gA = A + (size_t)(m0 + ldr) * K + ldc;       // + rr*64*K + k0
  const __bf16* gB = W + (size_t)(n0 + ldr) * K + ldc;

  // issue one stage of async tile DMA (4 x b128 per thread), bufsel static
  auto issue = [&](int k0, int bufsel) {
    #pragma unroll
    for (int rr = 0; rr < 2; ++rr) {
      async_ld_b128(dstA[bufsel][rr], gA + (size_t)rr * 64 * K + k0);
      async_ld_b128(dstB[bufsel][rr], gB + (size_t)rr * 64 * K + k0);
    }
  };

  // one K-step of WMMA work from LDS buffer `bufsel`
  auto compute = [&](int bufsel) {
    const __bf16* cA = lA + bufsel * LBUF;
    const __bf16* cB = lB + bufsel * LBUF;
    FragAB af[2], bfr[4];
    #pragma unroll
    for (int mt = 0; mt < 2; ++mt) {
      const int r = waveM * 32 + mt * 16 + col;
      af[mt].h[0] = *(const bf16x8*)(cA + r * LSTR + hl * 8);
      af[mt].h[1] = *(const bf16x8*)(cA + r * LSTR + 16 + hl * 8);
    }
    #pragma unroll
    for (int nt = 0; nt < 4; ++nt) {
      const int r = waveN * 64 + nt * 16 + col;
      bfr[nt].h[0] = *(const bf16x8*)(cB + r * LSTR + hl * 8);
      bfr[nt].h[1] = *(const bf16x8*)(cB + r * LSTR + 16 + hl * 8);
    }
    #pragma unroll
    for (int mt = 0; mt < 2; ++mt)
      #pragma unroll
      for (int nt = 0; nt < 4; ++nt)
        acc[mt][nt] = wmma_bf16(af[mt].v, bfr[nt].v, acc[mt][nt]);
  };

  // software pipeline: K assumed multiple of 2*BK (K=1024 here)
  issue(0, 0);
  int k0 = 0;
  for (; k0 + 2 * BK < K; k0 += 2 * BK) {
    issue(k0 + BK, 1);
    asm volatile("s_wait_asynccnt 0x4" ::: "memory");
    __syncthreads();
    compute(0);
    __syncthreads();
    issue(k0 + 2 * BK, 0);
    asm volatile("s_wait_asynccnt 0x4" ::: "memory");
    __syncthreads();
    compute(1);
    __syncthreads();
  }
  // tail: stages k0 (buf0) and k0+BK (buf1)
  issue(k0 + BK, 1);
  asm volatile("s_wait_asynccnt 0x4" ::: "memory");
  __syncthreads();
  compute(0);
  __syncthreads();
  asm volatile("s_wait_asynccnt 0x0" ::: "memory");
  __syncthreads();
  compute(1);

  // epilogue: bias add + store (C layout: lane col fixed, rows in elements)
  #pragma unroll
  for (int mt = 0; mt < 2; ++mt) {
    #pragma unroll
    for (int nt = 0; nt < 4; ++nt) {
      const int gn = n0 + waveN * 64 + nt * 16 + col;
      const float bvv = bias ? bias[gn] : 0.f;
      const int gm0 = m0 + waveM * 32 + mt * 16 + hl * 8;
      #pragma unroll
      for (int r = 0; r < 8; ++r) {
        const float v = acc[mt][nt][r] + bvv;
        const size_t idx = (size_t)(gm0 + r) * N + gn;
        if (outf) outf[idx] = v;
        else      outb[idx] = (__bf16)v;
      }
    }
  }
}

// ---------------------------------------------------------------------------
// Transpose V[b,s,h,d] (bf16) -> Vt[b,h,d,s] (bf16), 64x64 tiles via LDS
// ---------------------------------------------------------------------------
__global__ __launch_bounds__(256) void transpose_v(
    const __bf16* __restrict__ Vb, __bf16* __restrict__ Vt) {
  __shared__ __bf16 tile[64][65];
  const int s0 = blockIdx.x * 64;
  const int h = blockIdx.y, b = blockIdx.z;
  const int t = threadIdx.x;
  for (int i = t; i < 64 * 64; i += 256) {
    const int s = i >> 6, d = i & 63;
    tile[s][d] = Vb[((size_t)(b * S_LEN + s0 + s) * HEADS + h) * DHEAD + d];
  }
  __syncthreads();
  for (int i = t; i < 64 * 64; i += 256) {
    const int d = i >> 6, s = i & 63;
    Vt[((size_t)(b * HEADS + h) * DHEAD + d) * S_LEN + s0 + s] = tile[s][d];
  }
}

// ---------------------------------------------------------------------------
// Banded flash attention: one wave (32 lanes) per 16-query tile per (b,h).
// Skips key chunks outside the dilated band unless global rows/cols force
// them in. Scores + P@V via bf16 WMMA, fp32 online softmax.
// ---------------------------------------------------------------------------
__global__ __launch_bounds__(32) void attn_banded(
    const __bf16* __restrict__ Q, const __bf16* __restrict__ Kk,
    const __bf16* __restrict__ Vt, const unsigned* __restrict__ gmask,
    __bf16* __restrict__ O) {
  __shared__ __attribute__((aligned(16))) __bf16 ldsP[16 * 40];

  const int lane = threadIdx.x;
  const int col  = lane & 15;
  const int hl   = (lane >> 4) & 1;
  const int q0 = blockIdx.x * 16;
  const int h  = blockIdx.y, b = blockIdx.z;

  // Q fragments (A operand): rows q0..q0+15, d split 0:31 / 32:63
  FragAB qf[2];
  {
    const __bf16* qrow = Q + ((size_t)(b * S_LEN + q0 + col) * HEADS + h) * DHEAD;
    #pragma unroll
    for (int f = 0; f < 2; ++f) {
      qf[f].h[0] = *(const bf16x8*)(qrow + f * 32 + hl * 8);
      qf[f].h[1] = *(const bf16x8*)(qrow + f * 32 + 16 + hl * 8);
    }
  }

  float mrow[8], lrow[8];
  #pragma unroll
  for (int r = 0; r < 8; ++r) { mrow[r] = -__builtin_inff(); lrow[r] = 0.f; }
  v8f acc[4] = {};

  const unsigned qword = gmask[q0 >> 5];
  const bool anyqg = ((qword >> (q0 & 31)) & 0xFFFFu) != 0u;
  int lo = q0 - HALF_WIN;           if (lo < 0) lo = 0;
  int hi = q0 + 16 + HALF_WIN;      if (hi > S_LEN) hi = S_LEN;
  const int cLo = lo >> 5, cHi = (hi + 31) >> 5;

  for (int c = 0; c < NWORDS; ++c) {
    const unsigned kword = gmask[c];
    const bool inband = (c >= cLo) && (c < cHi);
    if (!inband && kword == 0u && !anyqg) continue;   // wave-uniform skip
    const int k0 = c << 5;

    // ---- scores: 16 queries x 32 keys (two 16x16 tiles) ----
    v8f sc[2];
    #pragma unroll
    for (int sub = 0; sub < 2; ++sub) {
      FragAB kf0, kf1;
      const __bf16* krow =
          Kk + ((size_t)(b * S_LEN + k0 + sub * 16 + col) * HEADS + h) * DHEAD;
      kf0.h[0] = *(const bf16x8*)(krow + hl * 8);
      kf0.h[1] = *(const bf16x8*)(krow + 16 + hl * 8);
      kf1.h[0] = *(const bf16x8*)(krow + 32 + hl * 8);
      kf1.h[1] = *(const bf16x8*)(krow + 48 + hl * 8);
      v8f z = {};
      z = wmma_bf16(qf[0].v, kf0.v, z);
      z = wmma_bf16(qf[1].v, kf1.v, z);
      sc[sub] = z;
    }

    // ---- scale + mask ----
    #pragma unroll
    for (int sub = 0; sub < 2; ++sub) {
      const int k = k0 + sub * 16 + col;
      const bool kg = ((kword >> (k & 31)) & 1u) != 0u;
      #pragma unroll
      for (int r = 0; r < 8; ++r) {
        const int q = q0 + r + hl * 8;
        const int d = q - k;
        const bool qg = ((qword >> (q & 31)) & 1u) != 0u;
        const bool ok =
            ((d >= -HALF_WIN) && (d <= HALF_WIN) && ((d & 1) == 0)) || qg || kg;
        const float v = sc[sub][r] * 0.125f;   // 1/sqrt(64)
        sc[sub][r] = ok ? v : -__builtin_inff();
      }
    }

    // ---- online softmax (row reductions over the 16 lanes of a half) ----
    #pragma unroll
    for (int r = 0; r < 8; ++r) {
      float mx = fmaxf(sc[0][r], sc[1][r]);
      mx = fmaxf(mx, __shfl_xor(mx, 1, 32));
      mx = fmaxf(mx, __shfl_xor(mx, 2, 32));
      mx = fmaxf(mx, __shfl_xor(mx, 4, 32));
      mx = fmaxf(mx, __shfl_xor(mx, 8, 32));
      const float newm = fmaxf(mrow[r], mx);
      const float safe = (newm == -__builtin_inff()) ? 0.f : newm;
      const float alpha = __expf(mrow[r] - safe);
      const float p0 = __expf(sc[0][r] - safe);
      const float p1 = __expf(sc[1][r] - safe);
      float rs = p0 + p1;
      rs += __shfl_xor(rs, 1, 32);
      rs += __shfl_xor(rs, 2, 32);
      rs += __shfl_xor(rs, 4, 32);
      rs += __shfl_xor(rs, 8, 32);
      lrow[r] = lrow[r] * alpha + rs;
      mrow[r] = newm;
      #pragma unroll
      for (int nt = 0; nt < 4; ++nt) acc[nt][r] = acc[nt][r] * alpha;
      sc[0][r] = p0;
      sc[1][r] = p1;
    }

    // ---- P: C-layout -> A-layout via LDS ----
    #pragma unroll
    for (int r = 0; r < 8; ++r) {
      ldsP[(r + hl * 8) * 40 + col]      = (__bf16)sc[0][r];
      ldsP[(r + hl * 8) * 40 + 16 + col] = (__bf16)sc[1][r];
    }
    asm volatile("s_wait_dscnt 0" ::: "memory");
    FragAB pf;
    pf.h[0] = *(const bf16x8*)(ldsP + col * 40 + hl * 8);
    pf.h[1] = *(const bf16x8*)(ldsP + col * 40 + 16 + hl * 8);
    asm volatile("s_wait_dscnt 0" ::: "memory");

    // ---- P @ V (Vt rows are contiguous in s) ----
    #pragma unroll
    for (int nt = 0; nt < 4; ++nt) {
      FragAB vf;
      const __bf16* vrow =
          Vt + ((size_t)(b * HEADS + h) * DHEAD + nt * 16 + col) * S_LEN + k0;
      vf.h[0] = *(const bf16x8*)(vrow + hl * 8);
      vf.h[1] = *(const bf16x8*)(vrow + 16 + hl * 8);
      acc[nt] = wmma_bf16(pf.v, vf.v, acc[nt]);
    }
  }

  // ---- normalize + store O[b,q,h*64+d] as bf16 ----
  #pragma unroll
  for (int nt = 0; nt < 4; ++nt) {
    #pragma unroll
    for (int r = 0; r < 8; ++r) {
      const float o = acc[nt][r] / lrow[r];
      O[((size_t)(b * S_LEN + q0 + r + hl * 8) * HEADS + h) * DHEAD + nt * 16 +
        col] = (__bf16)o;
    }
  }
}

// ---------------------------------------------------------------------------
// Host launch
// ---------------------------------------------------------------------------
extern "C" void kernel_launch(void* const* d_in, const int* in_sizes, int n_in,
                              void* d_out, int out_size, void* d_ws,
                              size_t ws_size, hipStream_t stream) {
  const float* x   = (const float*)d_in[0];
  const float* wq  = (const float*)d_in[1];
  const float* bq  = (const float*)d_in[2];
  const float* wv  = (const float*)d_in[3];
  const float* bv  = (const float*)d_in[4];
  const float* pq  = (const float*)d_in[5];
  const float* pbq = (const float*)d_in[6];
  const float* pk  = (const float*)d_in[7];
  const float* pbk = (const float*)d_in[8];
  const float* pv  = (const float*)d_in[9];
  const float* pbv = (const float*)d_in[10];
  const float* wo  = (const float*)d_in[11];
  const float* bo  = (const float*)d_in[12];
  const long long* gidx = (const long long*)d_in[13];

  char* ws = (char*)d_ws;
  const size_t ACT = (size_t)BATCH * S_LEN * DMODEL * sizeof(__bf16);  // 8 MB
  const size_t WT  = (size_t)DMODEL * DMODEL * sizeof(__bf16);         // 2 MB

  __bf16* xbf  = (__bf16*)(ws);                       // R0 (reused as Qbf)
  __bf16* wqb  = (__bf16*)(ws + ACT);
  __bf16* wvb  = (__bf16*)(ws + ACT + 1 * WT);
  __bf16* pqb  = (__bf16*)(ws + ACT + 2 * WT);
  __bf16* pkb  = (__bf16*)(ws + ACT + 3 * WT);
  __bf16* pvb  = (__bf16*)(ws + ACT + 4 * WT);
  __bf16* wob  = (__bf16*)(ws + ACT + 5 * WT);
  __bf16* q1   = (__bf16*)(ws + ACT + 6 * WT);        // R2 (reused as Obf)
  __bf16* v1   = (__bf16*)(ws + 2 * ACT + 6 * WT);    // R3 (reused as Vt)
  __bf16* kbuf = (__bf16*)(ws + 3 * ACT + 6 * WT);    // R4
  __bf16* vbuf = (__bf16*)(ws + 4 * ACT + 6 * WT);    // R5
  unsigned* gm = (unsigned*)(ws + 5 * ACT + 6 * WT);
  __bf16* qbf = xbf;   // xbf dead after first two GEMMs
  __bf16* vt  = v1;    // v1 dead after k/v in-projections
  __bf16* obf = q1;    // q1 dead after q in-projection

  const int M = BATCH * S_LEN;          // 4096
  const int nact4 = (M * DMODEL) / 4;   // 1M float4
  const int nwt4  = (DMODEL * DMODEL) / 4;

  // fp32 -> bf16
  cvt_f32_bf16<<<1024, 256, 0, stream>>>(x,  xbf, nact4);
  cvt_f32_bf16<<<256, 256, 0, stream>>>(wq, wqb, nwt4);
  cvt_f32_bf16<<<256, 256, 0, stream>>>(wv, wvb, nwt4);
  cvt_f32_bf16<<<256, 256, 0, stream>>>(pq, pqb, nwt4);
  cvt_f32_bf16<<<256, 256, 0, stream>>>(pk, pkb, nwt4);
  cvt_f32_bf16<<<256, 256, 0, stream>>>(pv, pvb, nwt4);
  cvt_f32_bf16<<<256, 256, 0, stream>>>(wo, wob, nwt4);
  build_gmask<<<1, 64, 0, stream>>>(gidx, gm, in_sizes[13]);

  dim3 gg(DMODEL / BN, M / BM);
  // q1 = x@wq^T + bq ; v1 = x@wv^T + bv (k shares v_proj, per reference)
  gemm_bf16<<<gg, 256, 0, stream>>>(xbf, wqb, bq, q1, nullptr, M, DMODEL, DMODEL);
  gemm_bf16<<<gg, 256, 0, stream>>>(xbf, wvb, bv, v1, nullptr, M, DMODEL, DMODEL);
  // head in-projections
  gemm_bf16<<<gg, 256, 0, stream>>>(q1, pqb, pbq, qbf,  nullptr, M, DMODEL, DMODEL);
  gemm_bf16<<<gg, 256, 0, stream>>>(v1, pkb, pbk, kbuf, nullptr, M, DMODEL, DMODEL);
  gemm_bf16<<<gg, 256, 0, stream>>>(v1, pvb, pbv, vbuf, nullptr, M, DMODEL, DMODEL);
  // V -> V^T for the P@V WMMA B-operand
  transpose_v<<<dim3(S_LEN / 64, HEADS, BATCH), 256, 0, stream>>>(vbuf, vt);
  // banded flash attention
  attn_banded<<<dim3(S_LEN / 16, HEADS, BATCH), 32, 0, stream>>>(qbf, kbuf, vt,
                                                                 gm, obf);
  // out projection (fp32 output)
  gemm_bf16<<<gg, 256, 0, stream>>>(obf, wob, bo, nullptr, (float*)d_out, M,
                                    DMODEL, DMODEL);
}
